// CirculantElmanCell_83236466196691
// MI455X (gfx1250) — matrix operand order repaired and checked by reading, not compiled, not run on hardware
//
#include <hip/hip_runtime.h>
#include <hip/hip_bf16.h>
#include <math.h>

// ---------------------------------------------------------------------------
// CirculantElmanCell on MI455X (gfx1250).
//   - all 3 matmuls via v_wmma_f32_16x16x32_bf16, dual accumulator chains
//   - B operands pre-swizzled to WMMA fragment layout (contiguous 32B/lane)
//   - A operands staged in LDS via global_load_async_to_lds_b128 (ASYNCcnt),
//     XOR-swizzled LDS addresses; CX init loads overlapped under the copy
//   - recurrence: persistent 16-block kernel, grid barrier per step, h kept
//     in a double-buffered bf16 staging buffer on the serial path
//   - epilogues: v_tanh_f32 / v_rcp_f32, non-temporal stream stores
// ---------------------------------------------------------------------------

#define DIM    2048
#define TSTEPS 1024
#define BATCH  16
#define TB     (TSTEPS * BATCH)       // 16384 rows for the parallel GEMMs
#define NB     (DIM / 16)             // 128 N-tiles
#define KC     (DIM / 32)             // 64 K-chunks of 32

typedef __attribute__((ext_vector_type(16))) __bf16 v16bf;
typedef __attribute__((ext_vector_type(8)))  __bf16 v8bf;
typedef __attribute__((ext_vector_type(4)))  __bf16 v4bf;
typedef __attribute__((ext_vector_type(8)))  float  v8f;

// workspace layout (bytes)
#define OFF_CNT  25165824ull                       // 24 MB: 3 bf16 matrices
#define OFF_HBF  (OFF_CNT + 4096ull)               // 2 x 16 x DIM bf16 (128 KB)
#define OFF_XBF  (OFF_HBF + 131072ull)             // x in bf16 (64 MB, optional)
#define WS_FULL  (OFF_XBF + (unsigned long long)TB * DIM * 2ull)

// float -> bf16, round-to-nearest-even
static __device__ inline __bf16 f2bf(float f) {
    unsigned u = __builtin_bit_cast(unsigned, f);
    unsigned r = u + 0x7FFFu + ((u >> 16) & 1u);
    unsigned short h = (unsigned short)(r >> 16);
    return __builtin_bit_cast(__bf16, h);
}

static __device__ inline float fast_tanh(float x) {
#if __has_builtin(__builtin_amdgcn_tanhf)
    return __builtin_amdgcn_tanhf(x);          // v_tanh_f32 (gfx1250 TRANS op)
#else
    float e = __expf(2.f * x);
    return 1.f - 2.f * __builtin_amdgcn_rcpf(e + 1.f);
#endif
}

static __device__ inline float fast_silu(float z) {
    return z * __builtin_amdgcn_rcpf(1.f + __expf(-z));   // v_rcp_f32
}

// LDS XOR swizzle: element (row, k) of a 16 x 2048 bf16 tile.
// 8-element (16B) chunks XORed by row -> conflict-free ds_load_b128 reads.
static __device__ inline int lds_idx(int row, int k) {
    int chunk = (k >> 3) ^ (row & 15);
    return row * DIM + (chunk << 3) + (k & 7);
}

// Issue async stage: copy a 16 x 2048 bf16 row-block into LDS with the XOR
// swizzle folded into the LDS destination address.  4096 chunks of 16B,
// lane-consecutive -> fully coalesced.  Tracked by ASYNCcnt (no wait here).
static __device__ inline void stage_async_issue(const __bf16* __restrict__ src,
                                                __bf16* smem) {
    unsigned lds_base = (unsigned)(unsigned long long)smem;  // low 32 = LDS addr
    #pragma unroll
    for (int i = 0; i < 16; ++i) {
        int c   = i * 256 + (int)threadIdx.x;   // chunk id, 0..4095
        int row = c >> 8;                       // 256 chunks per row
        int k8  = c & 255;
        const __bf16* g = src + ((size_t)row << 11) + ((size_t)k8 << 3);
        unsigned loff = (unsigned)(((row << 11) + ((k8 ^ (row & 15)) << 3)) * 2);
        unsigned laddr = lds_base + loff;
        asm volatile("global_load_async_to_lds_b128 %0, %1, off"
                     :: "v"(laddr), "v"((unsigned long long)g) : "memory");
    }
}

static __device__ inline void async_wait() {
    asm volatile("s_wait_asynccnt 0x0" ::: "memory");
}

// Sync fallback stage: f32 global -> cvt -> bf16 LDS (same swizzle).
static __device__ inline void stage_sync(const float* __restrict__ src,
                                         __bf16* smem) {
    for (int e = threadIdx.x; e < (16 * DIM) / 4; e += 256) {
        int row = e >> 9;
        int c4  = (e & 511) << 2;
        float4 v = *(const float4*)(src + (size_t)row * DIM + c4);
        int chunk = ((c4 >> 3) ^ (row & 15));
        __bf16* p = smem + row * DIM + (chunk << 3) + (c4 & 7);
        p[0] = f2bf(v.x); p[1] = f2bf(v.y); p[2] = f2bf(v.z); p[3] = f2bf(v.w);
    }
}

// A fragment (16x32 bf16) per ISA layout: lanes 0-15 take K=8g..8g+7 and
// K=16+8g..16+8g+7 -> two ds_load_b128.
static __device__ inline v16bf load_a(const __bf16* smem, int m, int kc, int g) {
    int k0 = kc * 32 + 8 * g;
    v8bf lo = *(const v8bf*)(smem + lds_idx(m, k0));
    v8bf hi = *(const v8bf*)(smem + lds_idx(m, k0 + 16));
    return __builtin_shufflevector(lo, hi, 0,1,2,3,4,5,6,7,8,9,10,11,12,13,14,15);
}

// B fragment: pre-swizzled so each lane's 16 bf16 are contiguous (2x b128).
static __device__ inline v16bf load_b(const __bf16* __restrict__ Bf, int kc,
                                      int nb, int lane) {
    return *(const v16bf*)(Bf + ((((size_t)kc * NB + nb) * 32 + lane) << 4));
}

// ---------------------------------------------------------------------------
// Kernel 1: build the three bf16 B-matrices directly in fragment layout.
// ---------------------------------------------------------------------------
__global__ void __launch_bounds__(256) build_frags(
    const float* __restrict__ W, const float* __restrict__ cx,
    const float* __restrict__ ch,
    __bf16* __restrict__ Bg, __bf16* __restrict__ Bcx, __bf16* __restrict__ Bch)
{
    unsigned tid = blockIdx.x * 256u + threadIdx.x;        // 3 * 2^22 total
    unsigned mat = tid >> 22;
    unsigned idx = tid & 0x3FFFFFu;
    unsigned j    = idx & 15u;
    unsigned lane = (idx >> 4) & 31u;
    unsigned nbkc = idx >> 9;
    unsigned nb   = nbkc & (NB - 1);
    unsigned kc   = nbkc >> 7;
    unsigned K = kc * 32u + ((lane >> 4) << 3) + (j & 7u) + ((j >> 3) << 4);
    unsigned n = (nb << 4) + (lane & 15u);
    float v;
    if (mat == 0)      v = W[(size_t)n * DIM + K];
    else if (mat == 1) v = cx[(n - K) & (DIM - 1)];
    else               v = ch[(n - K) & (DIM - 1)];
    __bf16 bv = f2bf(v);
    if (mat == 0)      Bg[idx]  = bv;
    else if (mat == 1) Bcx[idx] = bv;
    else               Bch[idx] = bv;
}

// ---------------------------------------------------------------------------
// Kernel 2: f32 -> bf16 bulk convert (x pre-conversion and h0 staging init).
// ---------------------------------------------------------------------------
__global__ void __launch_bounds__(256) convert_bf16(
    const float* __restrict__ src, __bf16* __restrict__ dst, int n4)
{
    int i = blockIdx.x * 256 + threadIdx.x;
    if (i < n4) {
        float4 v = *(const float4*)(src + 4 * (size_t)i);
        v4bf o;
        o[0] = f2bf(v.x); o[1] = f2bf(v.y); o[2] = f2bf(v.z); o[3] = f2bf(v.w);
        *(v4bf*)(dst + 4 * (size_t)i) = o;
    }
}

// ---------------------------------------------------------------------------
// Kernel 3: C[r][n] = act( x[r][:] . Bf[:][n] + bias[n] ) for 16384 rows.
// Grid (TB/16, 4); 8 waves x 4 N-tiles each.  Abf != null -> async bf16 stage.
// ---------------------------------------------------------------------------
__global__ void __launch_bounds__(256) gemm16(
    const float* __restrict__ A, const __bf16* __restrict__ Abf,
    const __bf16* __restrict__ Bf, const float* __restrict__ bias,
    float* __restrict__ outp, int act)
{
    extern __shared__ __bf16 smem[];                 // 16*2048 bf16 = 64KB
    const int rowBase = blockIdx.x * 16;
    if (Abf) { stage_async_issue(Abf + (size_t)rowBase * DIM, smem); async_wait(); }
    else     { stage_sync(A + (size_t)rowBase * DIM, smem); }
    __syncthreads();

    const int lane = threadIdx.x & 31;
    const int wave = threadIdx.x >> 5;
    const int g = lane >> 4;
    const int m = lane & 15;

    for (int tsub = 0; tsub < 4; ++tsub) {
        const int nb = blockIdx.y * 32 + wave * 4 + tsub;
        v8f acc0 = {}, acc1 = {};                    // 2 independent chains
        #pragma unroll 2
        for (int kc = 0; kc < KC; kc += 2) {
            v16bf a0 = load_a(smem, m, kc, g);
            v16bf b0 = load_b(Bf, kc, nb, lane);
            acc0 = __builtin_amdgcn_wmma_f32_16x16x32_bf16(
                       false, a0, false, b0, (short)0, acc0, false, false);
            v16bf a1 = load_a(smem, m, kc + 1, g);
            v16bf b1 = load_b(Bf, kc + 1, nb, lane);
            acc1 = __builtin_amdgcn_wmma_f32_16x16x32_bf16(
                       false, a1, false, b1, (short)0, acc1, false, false);
        }
        const int col = nb * 16 + m;
        const float bb = bias[col];
        #pragma unroll
        for (int i = 0; i < 8; ++i) {
            int row = rowBase + i + 8 * g;             // C layout: M = i + 8*g
            float z = acc0[i] + acc1[i] + bb;
            if (act) z = fast_silu(z);
            __builtin_nontemporal_store(z, outp + (size_t)row * DIM + col);
        }
    }
}

// ---------------------------------------------------------------------------
// Kernel 4: persistent recurrence.  16 blocks x 8 waves = 128 N-tiles.
//   issue async stage of h_{t-1} (bf16, double-buffered) -> LDS
//   overlap: acc0 = CX_t (pre-stored in h slot t+1) under the copy
//   acc += h_{t-1} x circ(c_h) via WMMA (2 chains)
//   h_t = tanh(acc); out_t = h_t * siluG_t; emit bf16 h_t for next step
// ---------------------------------------------------------------------------
__global__ void __launch_bounds__(256) step_kernel(
    const __bf16* __restrict__ Bch, float* __restrict__ outr,
    float* __restrict__ hseq, __bf16* __restrict__ hbf,
    unsigned* __restrict__ counter)
{
    extern __shared__ __bf16 smem[];
    const int lane = threadIdx.x & 31;
    const int wave = threadIdx.x >> 5;
    const int g = lane >> 4;
    const int m = lane & 15;
    const int nb  = blockIdx.x * 8 + wave;
    const int col = nb * 16 + m;
    const unsigned nblk = gridDim.x;

    for (int t = 0; t < TSTEPS; ++t) {
        const __bf16* src = hbf + (size_t)(t & 1)       * (BATCH * DIM);
        __bf16*       dst = hbf + (size_t)((t + 1) & 1) * (BATCH * DIM);
        float*      hnext = hseq + (size_t)(t + 1) * (BATCH * DIM);
        float*       outt = outr + (size_t)t       * (BATCH * DIM);

        stage_async_issue(src, smem);    // ASYNCcnt copy in flight ...

        v8f acc0, acc1 = {};
        #pragma unroll
        for (int i = 0; i < 8; ++i)      // ... overlap CX_t + b loads under it
            acc0[i] = __builtin_nontemporal_load(
                          hnext + (size_t)(i + 8 * g) * DIM + col);

        async_wait();                    // s_wait_asynccnt 0
        __syncthreads();

        #pragma unroll 2
        for (int kc = 0; kc < KC; kc += 2) {
            v16bf a0 = load_a(smem, m, kc, g);
            v16bf b0 = load_b(Bch, kc, nb, lane);
            acc0 = __builtin_amdgcn_wmma_f32_16x16x32_bf16(
                       false, a0, false, b0, (short)0, acc0, false, false);
            v16bf a1 = load_a(smem, m, kc + 1, g);
            v16bf b1 = load_b(Bch, kc + 1, nb, lane);
            acc1 = __builtin_amdgcn_wmma_f32_16x16x32_bf16(
                       false, a1, false, b1, (short)0, acc1, false, false);
        }

        #pragma unroll
        for (int i = 0; i < 8; ++i) {
            int row = i + 8 * g;
            size_t idx = (size_t)row * DIM + col;
            float hn = fast_tanh(acc0[i] + acc1[i]);       // v_tanh_f32
            __builtin_nontemporal_store(hn, hnext + idx);  // f32 h: output only
            float gv = __builtin_nontemporal_load(outt + idx);
            __builtin_nontemporal_store(hn * gv, outt + idx);
            dst[idx] = f2bf(hn);                           // bf16 h, next step
        }

        // device-wide barrier before next step reads dst / hbf
        __threadfence();
        __syncthreads();
        if (threadIdx.x == 0) {
            __hip_atomic_fetch_add(counter, 1u, __ATOMIC_ACQ_REL,
                                   __HIP_MEMORY_SCOPE_AGENT);
            const unsigned target = nblk * (unsigned)(t + 1);
            while (__hip_atomic_load(counter, __ATOMIC_ACQUIRE,
                                     __HIP_MEMORY_SCOPE_AGENT) < target) {
                __builtin_amdgcn_s_sleep(2);
            }
        }
        __syncthreads();
    }
}

// ---------------------------------------------------------------------------
extern "C" void kernel_launch(void* const* d_in, const int* in_sizes, int n_in,
                              void* d_out, int out_size, void* d_ws, size_t ws_size,
                              hipStream_t stream) {
    (void)in_sizes; (void)n_in; (void)out_size;
    const float* x      = (const float*)d_in[0];
    const float* h0     = (const float*)d_in[1];
    const float* c_h    = (const float*)d_in[2];
    const float* c_x    = (const float*)d_in[3];
    const float* W_gate = (const float*)d_in[4];
    const float* b      = (const float*)d_in[5];
    const float* b_gate = (const float*)d_in[6];

    float* out  = (float*)d_out;                         // [T,B,D]
    float* hseq = out + (size_t)TSTEPS * BATCH * DIM;    // [T+1,B,D]

    const size_t MATE = (size_t)DIM * DIM;               // elems per matrix
    __bf16* Bg  = (__bf16*)d_ws;
    __bf16* Bcx = Bg + MATE;
    __bf16* Bch = Bcx + MATE;
    unsigned* counter = (unsigned*)((char*)d_ws + OFF_CNT);
    __bf16* hbf = (__bf16*)((char*)d_ws + OFF_HBF);      // 2 x [16,D] bf16
    __bf16* xbf = (__bf16*)((char*)d_ws + OFF_XBF);      // [TB,D] bf16 (optional)
    const bool full = ws_size >= (size_t)WS_FULL;

    hipMemsetAsync(counter, 0, 64, stream);

    // h[0] = h0 (f32 output) and hbf[0] = bf16(h0) (recurrence staging)
    hipMemcpyAsync(hseq, h0, (size_t)BATCH * DIM * sizeof(float),
                   hipMemcpyDeviceToDevice, stream);
    convert_bf16<<<(BATCH * DIM / 4 + 255) / 256, 256, 0, stream>>>(
        h0, hbf, BATCH * DIM / 4);

    // Build the three fragment-layout bf16 matrices (24 MB).
    build_frags<<<(3u * (unsigned)MATE) / 256u, 256, 0, stream>>>(
        W_gate, c_x, c_h, Bg, Bcx, Bch);

    // Optional: pre-convert x to bf16 so the big GEMMs stage via async copies.
    if (full)
        convert_bf16<<<(TB * DIM / 4 + 255) / 256, 256, 0, stream>>>(
            x, xbf, TB * DIM / 4);
    const __bf16* Abf = full ? xbf : (const __bf16*)nullptr;

    // Phase 1: silu(gate) -> output region, circ_x + b -> h slots 1..T.
    dim3 g1(TB / 16, 4);
    gemm16<<<g1, 256, 16 * DIM * sizeof(__bf16), stream>>>(x, Abf, Bg,  b_gate,
                                                           out, 1);
    gemm16<<<g1, 256, 16 * DIM * sizeof(__bf16), stream>>>(x, Abf, Bcx, b,
                                                           hseq + (size_t)BATCH * DIM, 0);

    // Phase 2: persistent recurrence over 1024 steps.
    step_kernel<<<16, 256, 16 * DIM * sizeof(__bf16), stream>>>(Bch, out, hseq,
                                                                hbf, counter);
}
